// DecoderGRU_70703751627350
// MI455X (gfx1250) — matrix operand order repaired and checked by reading, not compiled, not run on hardware
//
#include <hip/hip_runtime.h>
#include <hip/hip_bf16.h>

typedef __attribute__((ext_vector_type(16))) _Float16 v16h;
typedef __attribute__((ext_vector_type(8)))  float    v8f;

#define NEGINF (-1e30f)

// Problem constants
constexpr int kV = 50257, kE = 256, kH = 512, kM = 512;
constexpr int kB = 8, kT = 64, kU = 64, kPER = 10, kW = kU * kPER; // 640

// ---------------------------------------------------------------------------
// Generic WMMA GEMM:  C[m,n] = sum_k A[m,k] * Wt[n,k]  (+ bias[n])
// CONTRACT: lda == Kdim (A row panel is contiguous) — true for all call sites.
// A fp32 [Mdim x K], Wt fp32 [Ndim x K], f16-converted in registers, fp32
// accumulate via v_wmma_f32_16x16x32_f16.
// Block = 128 threads (4 waves) sharing one 16-row M tile; the A panel is
// staged block-wide into LDS with GLOBAL_LOAD_ASYNC_TO_LDS_B128 (ASYNCcnt),
// then each wave computes 4 N tiles. blockIdx.z = batch.
// Dynamic LDS: 16 * Kdim * sizeof(float).
// ---------------------------------------------------------------------------
__global__ __launch_bounds__(128)
void wmma_gemm_kernel(const float* __restrict__ A, long lda, long strideAb,
                      const float* __restrict__ Wt, long ldw, long strideWb,
                      const float* __restrict__ bias,
                      float* __restrict__ C, long ldc, long strideCb,
                      int Mdim, int Ndim, int Kdim)
{
    extern __shared__ __align__(16) float apanel[];   // [16][Kdim] fp32

    const int lane = threadIdx.x & 31;
    const int wave = threadIdx.x >> 5;        // 0..3
    const int half = lane >> 4;               // 0 or 1
    const int l16  = lane & 15;

    A  += (long)blockIdx.z * strideAb;
    Wt += (long)blockIdx.z * strideWb;
    C  += (long)blockIdx.z * strideCb;

    const int m0 = blockIdx.y * 16;
    if (m0 >= Mdim) return;                   // uniform -> EXEC stays full

    // ---- Async-stage the 16 x Kdim A panel into LDS (16B chunks) ----
    // 4*Kdim chunks, Kdim in {512,640,1024} -> multiple of 128, no divergence.
    {
        const float* apanel_g = A + (long)m0 * lda;
        const int nchunks = 4 * Kdim;
        for (int c = threadIdx.x; c < nchunks; c += 128) {
            const float* gp = apanel_g + (long)c * 4;
            unsigned int loff = (unsigned int)(size_t)(&apanel[c * 4]);
            asm volatile("global_load_async_to_lds_b128 %0, %1, off"
                         :: "v"(loff), "v"(gp) : "memory");
        }
        asm volatile("s_wait_asynccnt 0x0" ::: "memory");
        __syncthreads();
    }

    const int ncol0 = (blockIdx.x * 4 + wave) * 64;   // first column of this wave

    v8f acc[4];
#pragma unroll
    for (int j = 0; j < 4; ++j)
#pragma unroll
        for (int i = 0; i < 8; ++i) acc[j][i] = 0.0f;

    const float* arow = apanel + (long)l16 * Kdim;    // this lane's A row (LDS)
    const int kAoff = half * 8;

    for (int kk = 0; kk < Kdim; kk += 32) {
        // ---- A fragment: 16x32 f16 per ISA layout; K pattern is two
        //      aligned float4 pairs: {0..7} and {16..23} (+8 for upper half)
        v16h a;
        {
            const float* ap = arow + kk + kAoff;
            const float4 f0 = *(const float4*)(ap + 0);
            const float4 f1 = *(const float4*)(ap + 4);
            const float4 f2 = *(const float4*)(ap + 16);
            const float4 f3 = *(const float4*)(ap + 20);
            a[0]  = (_Float16)f0.x; a[1]  = (_Float16)f0.y;
            a[2]  = (_Float16)f0.z; a[3]  = (_Float16)f0.w;
            a[4]  = (_Float16)f1.x; a[5]  = (_Float16)f1.y;
            a[6]  = (_Float16)f1.z; a[7]  = (_Float16)f1.w;
            a[8]  = (_Float16)f2.x; a[9]  = (_Float16)f2.y;
            a[10] = (_Float16)f2.z; a[11] = (_Float16)f2.w;
            a[12] = (_Float16)f3.x; a[13] = (_Float16)f3.y;
            a[14] = (_Float16)f3.z; a[15] = (_Float16)f3.w;
        }
        // ---- 4 B fragments (32x16 f16), WMMA into 4 accumulators ----
#pragma unroll
        for (int j = 0; j < 4; ++j) {
            int col  = ncol0 + j * 16 + l16;
            int colc = col < Ndim ? col : (Ndim - 1);          // clamp loads, mask stores
            const float* wrow = Wt + (long)colc * ldw + kk + half * 8;
            if (kk + 32 < Kdim) __builtin_prefetch(wrow + 32, 0, 1);
            v16h bf;
#pragma unroll
            for (int q = 0; q < 4; ++q) {
                const float4 f = *(const float4*)(wrow + 4 * q + (q >> 1) * 8);
                bf[4 * q + 0] = (_Float16)f.x;
                bf[4 * q + 1] = (_Float16)f.y;
                bf[4 * q + 2] = (_Float16)f.z;
                bf[4 * q + 3] = (_Float16)f.w;
            }
            acc[j] = __builtin_amdgcn_wmma_f32_16x16x32_f16(
                false, a, false, bf, (short)0, acc[j], false, false);
        }
    }

#pragma unroll
    for (int j = 0; j < 4; ++j) {
        const int col = ncol0 + j * 16 + l16;
        if (col < Ndim) {
            const float bv = bias ? bias[col] : 0.0f;
#pragma unroll
            for (int i = 0; i < 8; ++i) {
                const int row = m0 + i + half * 8;
                C[(long)row * ldc + col] = acc[j][i] + bv;
            }
        }
    }
}

// ---------------------------------------------------------------------------
// Embedding lookup: x_emb[b,t,:] = emb[target[b,t], :]
// ---------------------------------------------------------------------------
__global__ __launch_bounds__(256)
void embed_kernel(const int* __restrict__ target, const float* __restrict__ emb,
                  float* __restrict__ x_emb)
{
    const int bt = blockIdx.x;            // 0..511
    const int e  = threadIdx.x;           // 0..255
    const long tok = target[bt];
    x_emb[(long)bt * kE + e] = emb[tok * kE + e];
}

// h0 = u_output[b, u_len[b]-1, :] replicated into both layers' state buffers
__global__ __launch_bounds__(256)
void init_h_kernel(const int* __restrict__ u_len, const float* __restrict__ u_output,
                   float* __restrict__ hA0, float* __restrict__ hB0)
{
    const int b = blockIdx.x;
    const int idx = u_len[b] - 1;
    const float* src = u_output + ((long)b * kU + idx) * kM;
    for (int m = threadIdx.x; m < kH; m += 256) {
        const float v = src[m];
        hA0[b * kH + m] = v;
        hB0[b * kH + m] = v;
    }
}

// ---------------------------------------------------------------------------
// One GRU step for one layer (torch gate math). 4096 threads total: (b,j).
// x_t and h_prev staged in LDS; weight rows streamed as float4.
// ---------------------------------------------------------------------------
__global__ __launch_bounds__(256)
void gru_step_kernel(const float* __restrict__ x, long x_bstride, int Din,
                     const float* __restrict__ hprev, float* __restrict__ hnew,
                     float* __restrict__ out, long out_bstride,
                     const float* __restrict__ Wih, const float* __restrict__ Whh,
                     const float* __restrict__ bih, const float* __restrict__ bhh)
{
    __shared__ __align__(16) float xs[kH];
    __shared__ __align__(16) float hs[kH];
    const int tid = threadIdx.x;
    const int gid = blockIdx.x * 256 + tid;   // 0..4095
    const int b = gid >> 9;                   // /512 (one b per block pair)
    const int j = gid & (kH - 1);

    const float* xb = x + (long)b * x_bstride;
    for (int k = tid; k < Din; k += 256) xs[k] = xb[k];
    const float* hb = hprev + b * kH;
    for (int k = tid; k < kH; k += 256) hs[k] = hb[k];
    __syncthreads();

    float gi[3], gh[3];
#pragma unroll
    for (int g = 0; g < 3; ++g) {
        const float* wi = Wih + (long)(g * kH + j) * Din;
        float s = bih[g * kH + j];
        for (int k = 0; k < Din; k += 4) {
            const float4 w4 = *(const float4*)(wi + k);
            const float4 x4 = *(const float4*)(xs + k);
            s += w4.x * x4.x + w4.y * x4.y + w4.z * x4.z + w4.w * x4.w;
        }
        gi[g] = s;
        const float* wh = Whh + (long)(g * kH + j) * kH;
        float t = bhh[g * kH + j];
        for (int k = 0; k < kH; k += 4) {
            const float4 w4 = *(const float4*)(wh + k);
            const float4 h4 = *(const float4*)(hs + k);
            t += w4.x * h4.x + w4.y * h4.y + w4.z * h4.z + w4.w * h4.w;
        }
        gh[g] = t;
    }
    const float r = 1.0f / (1.0f + __expf(-(gi[0] + gh[0])));
    const float z = 1.0f / (1.0f + __expf(-(gi[1] + gh[1])));
    const float n = tanhf(gi[2] + r * gh[2]);
    const float hv = (1.0f - z) * n + z * hs[j];
    hnew[b * kH + j] = hv;
    out[(long)b * out_bstride + j] = hv;
}

// wT[b,m,w] = w_output[b,w,m]
__global__ __launch_bounds__(256)
void transpose_w_kernel(const float* __restrict__ w_output, float* __restrict__ wT)
{
    const long i = (long)blockIdx.x * 256 + threadIdx.x;
    if (i >= (long)kB * kM * kW) return;
    const int w = (int)(i % kW);
    const long r = i / kW;
    const int m = (int)(r % kM);
    const int b = (int)(r / kM);
    wT[i] = w_output[((long)b * kW + w) * kM + m];
}

// log_softmax over U with length mask; also emit exp(logsm_u) output
__global__ __launch_bounds__(64)
void softmax_u_kernel(const float* __restrict__ su, const int* __restrict__ u_len,
                      float* __restrict__ logsm_u, float* __restrict__ pu_out)
{
    const int bt = blockIdx.x;            // 0..511
    const int b  = bt >> 6;
    const int u  = threadIdx.x;           // 0..63
    __shared__ float red[kU];
    float s = su[(long)bt * kU + u];
    if (u >= u_len[b]) s = NEGINF;
    red[u] = s; __syncthreads();
    for (int off = 32; off; off >>= 1) {
        if (u < off) red[u] = fmaxf(red[u], red[u + off]);
        __syncthreads();
    }
    const float mx = red[0]; __syncthreads();
    red[u] = __expf(s - mx); __syncthreads();
    for (int off = 32; off; off >>= 1) {
        if (u < off) red[u] += red[u + off];
        __syncthreads();
    }
    const float ls = s - mx - __logf(red[0]);
    logsm_u[(long)bt * kU + u] = ls;
    pu_out[(long)bt * kU + u]  = __expf(ls);
}

// Per-utterance segment softmax over words + combine with logsm_u
__global__ __launch_bounds__(640)
void seg_softmax_kernel(const float* __restrict__ swb, const int* __restrict__ w_len,
                        const int* __restrict__ seg_id, const float* __restrict__ logsm_u,
                        float* __restrict__ suw_out)
{
    const int bt = blockIdx.x;            // 0..511
    const int b  = bt >> 6;
    const int w  = threadIdx.x;           // 0..639
    __shared__ float s[kW];
    __shared__ int   sid[kW];
    __shared__ float gmx[kU];
    __shared__ float gsum[kU];

    const int wl = w_len[b];
    float v = swb[(long)bt * kW + w];
    if (w >= wl) v = NEGINF;
    s[w]   = v;
    sid[w] = seg_id[b * kW + w];
    __syncthreads();

    if (w < kU) {
        float mx = NEGINF;
        for (int k = 0; k < kW; ++k)
            if (sid[k] == w) mx = fmaxf(mx, s[k]);
        gmx[w] = mx;
    }
    __syncthreads();
    if (w < kU) {
        const float mx = gmx[w];
        float sum = 0.0f;
        for (int k = 0; k < kW; ++k)
            if (sid[k] == w) sum += __expf(s[k] - mx);
        gsum[w] = sum;
    }
    __syncthreads();

    const int g = sid[w];
    const float lsw = v - gmx[g] - __logf(gsum[g]);
    const float lu  = logsm_u[(long)bt * kU + g];
    float outv = __expf(lu + lsw);
    if (w >= wl) outv = 0.0f;
    suw_out[(long)bt * kW + w] = outv;
}

// cat = [context | rnn] per (b,t) row -> [512, 1024]
__global__ __launch_bounds__(256)
void cat_kernel(const float* __restrict__ context, const float* __restrict__ rnn,
                float* __restrict__ cat)
{
    const long i = (long)blockIdx.x * 256 + threadIdx.x;
    if (i >= (long)kB * kT * (kM + kH)) return;
    const int  c   = (int)(i & 1023);
    const long row = i >> 10;
    cat[i] = (c < kM) ? context[row * kM + c] : rnn[row * kH + (c - kM)];
}

// In-place log_softmax over V per row of dec
__global__ __launch_bounds__(256)
void logsoftmax_v_kernel(float* __restrict__ dec)
{
    const int row = blockIdx.x;           // 0..511
    const int tid = threadIdx.x;
    float* p = dec + (long)row * kV;
    __shared__ float red[256];

    float mx = NEGINF;
    for (int i = tid; i < kV; i += 256) mx = fmaxf(mx, p[i]);
    red[tid] = mx; __syncthreads();
    for (int off = 128; off; off >>= 1) {
        if (tid < off) red[tid] = fmaxf(red[tid], red[tid + off]);
        __syncthreads();
    }
    mx = red[0]; __syncthreads();

    float sum = 0.0f;
    for (int i = tid; i < kV; i += 256) sum += __expf(p[i] - mx);
    red[tid] = sum; __syncthreads();
    for (int off = 128; off; off >>= 1) {
        if (tid < off) red[tid] += red[tid + off];
        __syncthreads();
    }
    const float lse = mx + __logf(red[0]);
    for (int i = tid; i < kV; i += 256) p[i] = p[i] - lse;
}

// ---------------------------------------------------------------------------
extern "C" void kernel_launch(void* const* d_in, const int* in_sizes, int n_in,
                              void* d_out, int out_size, void* d_ws, size_t ws_size,
                              hipStream_t stream)
{
    (void)in_sizes; (void)n_in; (void)out_size; (void)ws_size;

    const int*   target   = (const int*)  d_in[0];
    const int*   u_len    = (const int*)  d_in[1];
    const int*   w_len    = (const int*)  d_in[2];
    const int*   seg_id   = (const int*)  d_in[3];
    const float* u_output = (const float*)d_in[4];
    const float* w_output = (const float*)d_in[5];
    const float* emb      = (const float*)d_in[6];
    const float* W_ih0    = (const float*)d_in[7];
    const float* W_hh0    = (const float*)d_in[8];
    const float* b_ih0    = (const float*)d_in[9];
    const float* b_hh0    = (const float*)d_in[10];
    const float* W_ih1    = (const float*)d_in[11];
    const float* W_hh1    = (const float*)d_in[12];
    const float* b_ih1    = (const float*)d_in[13];
    const float* b_hh1    = (const float*)d_in[14];
    const float* Wu       = (const float*)d_in[15];
    const float* bu       = (const float*)d_in[16];
    const float* Ww       = (const float*)d_in[17];
    const float* bw       = (const float*)d_in[18];
    const float* Wo       = (const float*)d_in[19];
    const float* bo       = (const float*)d_in[20];

    float* out = (float*)d_out;
    float* dec = out;                                    // [B,T,V]
    float* suw = out + (long)kB * kT * kV;               // [B,T,W]
    float* pu  = suw + (long)kB * kT * kW;               // [B,T,U]

    // Workspace carve-up (floats)
    float* ws    = (float*)d_ws;
    float* x_emb = ws;                         // 131072   [B,T,E]
    float* mid   = x_emb + (long)kB * kT * kE; // 262144   [B,T,H] layer0 out
    float* rnn   = mid   + (long)kB * kT * kH; // 262144   [B,T,H] layer1 out
    float* hA    = rnn   + (long)kB * kT * kH; // 2 x 4096 layer0 state (double buffer)
    float* hB    = hA    + 2L * kB * kH;       // 2 x 4096 layer1 state
    float* au    = hB    + 2L * kB * kH;       // 262144   [B,U,H]
    float* aw    = au    + (long)kB * kU * kH; // 2621440  [B,W,H]
    float* wT    = aw    + (long)kB * kW * kH; // 2621440  [B,M,W]
    float* su    = wT    + (long)kB * kM * kW; // 32768    [B,T,U]
    float* lsu   = su    + (long)kB * kT * kU; // 32768    [B,T,U]
    float* swb   = lsu   + (long)kB * kT * kU; // 327680   [B,T,W]
    float* ctx   = swb   + (long)kB * kT * kW; // 262144   [B,T,M]
    float* cat   = ctx   + (long)kB * kT * kM; // 524288   [B,T,H+M]

    // 1) Embedding + initial hidden state
    embed_kernel<<<kB * kT, 256, 0, stream>>>(target, emb, x_emb);
    init_h_kernel<<<kB, 256, 0, stream>>>(u_len, u_output, hA, hB);

    // 2) Sequential 2-layer GRU (one launch per step per layer; graph-replayed)
    for (int t = 0; t < kT; ++t) {
        const int cur = t & 1, nxt = (t + 1) & 1;
        gru_step_kernel<<<16, 256, 0, stream>>>(
            x_emb + (long)t * kE, (long)kT * kE, kE,
            hA + (long)cur * kB * kH, hA + (long)nxt * kB * kH,
            mid + (long)t * kH, (long)kT * kH,
            W_ih0, W_hh0, b_ih0, b_hh0);
        gru_step_kernel<<<16, 256, 0, stream>>>(
            mid + (long)t * kH, (long)kT * kH, kH,
            hB + (long)cur * kB * kH, hB + (long)nxt * kB * kH,
            rnn + (long)t * kH, (long)kT * kH,
            W_ih1, W_hh1, b_ih1, b_hh1);
    }

    const size_t shK512  = 16 * 512  * sizeof(float);   // 32 KB
    const size_t shK640  = 16 * 640  * sizeof(float);   // 40 KB
    const size_t shK1024 = 16 * 1024 * sizeof(float);   // 64 KB

    // 3) au = u_output @ Wu^T + bu   (M=512, N=512, K=512)
    wmma_gemm_kernel<<<dim3(2, 32, 1), 128, shK512, stream>>>(
        u_output, kM, 0, Wu, kM, 0, bu, au, kH, 0, kB * kU, kH, kM);

    // 4) aw = w_output @ Ww^T + bw   (M=5120, N=512, K=512)
    wmma_gemm_kernel<<<dim3(2, 320, 1), 128, shK512, stream>>>(
        w_output, kM, 0, Ww, kM, 0, bw, aw, kH, 0, kB * kW, kH, kM);

    // 5) wT[b] = w_output[b]^T
    transpose_w_kernel<<<(kB * kM * kW + 255) / 256, 256, 0, stream>>>(w_output, wT);

    // 6) scores_u[b] = rnn[b] @ au[b]^T  (batched: M=64, N=64, K=512)
    wmma_gemm_kernel<<<dim3(1, 4, kB), 128, shK512, stream>>>(
        rnn, kH, (long)kT * kH, au, kH, (long)kU * kH, nullptr,
        su, kU, (long)kT * kU, kT, kU, kH);

    // 7) masked log_softmax over U; emit exp() output
    softmax_u_kernel<<<kB * kT, 64, 0, stream>>>(su, u_len, lsu, pu);

    // 8) sw[b] = rnn[b] @ aw[b]^T  (batched: M=64, N=640, K=512)
    wmma_gemm_kernel<<<dim3(3, 4, kB), 128, shK512, stream>>>(
        rnn, kH, (long)kT * kH, aw, kH, (long)kW * kH, nullptr,
        swb, kW, (long)kT * kW, kT, kW, kH);

    // 9) segment softmax + combine -> scores_uw (output #2, reused as GEMM A)
    seg_softmax_kernel<<<kB * kT, kW, 0, stream>>>(swb, w_len, seg_id, lsu, suw);

    // 10) context[b] = scores_uw[b] @ w_output[b]  (M=64, N=512, K=640)
    wmma_gemm_kernel<<<dim3(2, 4, kB), 128, shK640, stream>>>(
        suw, kW, (long)kT * kW, wT, kW, (long)kM * kW, nullptr,
        ctx, kM, (long)kT * kM, kT, kM, kW);

    // 11) cat = [context | rnn]
    cat_kernel<<<(kB * kT * (kM + kH) + 255) / 256, 256, 0, stream>>>(ctx, rnn, cat);

    // 12) dec = cat @ Wo^T + bo  (M=512, N=50257, K=1024) — the big WMMA GEMM
    wmma_gemm_kernel<<<dim3((kV + 255) / 256, 32, 1), 128, shK1024, stream>>>(
        cat, kH + kM, 0, Wo, kH + kM, 0, bo, dec, kV, 0,
        kB * kT, kV, kH + kM);

    // 13) in-place log_softmax over V
    logsoftmax_v_kernel<<<kB * kT, 256, 0, stream>>>(dec);
}
